// MoEBlock_48533130445599
// MI455X (gfx1250) — compile-verified, matrix operand order checked by reading, not run on hardware
//
#include <hip/hip_runtime.h>

// ---------------- types ----------------
typedef __attribute__((ext_vector_type(8)))  _Float16 v8h;
typedef __attribute__((ext_vector_type(16))) _Float16 v16h;
typedef __attribute__((ext_vector_type(8)))  float    v8f;
typedef __attribute__((ext_vector_type(4)))  int      v4i;

#define NEXP 6
#define CCH  256
#define IMGW 64
#define IMGH 64
#define NPIX 4096              // 64*64
#define NB   16
#define OUT_ELEMS (16777216)   // 16*256*64*64
#define WELEMS (6*9*256*256)   // per-layer transposed f16 weight elements

#define A_STRIDE 40            // halves per A row (32 used + 8 pad) -> 80B, bank-rotating
#define B_STRIDE 40            // halves per B position (32 cin + 8 pad)
#define B_POS    264           // 4 rows x 66 cols halo tile positions

// ---------------- optional gfx1250 async global->LDS path ----------------
#if __has_builtin(__builtin_amdgcn_global_load_async_to_lds_b128) && __has_builtin(__builtin_amdgcn_s_wait_asynccnt)
#define HAS_ASYNC_LDS 1
typedef __attribute__((address_space(1))) v4i gv4i_t;   // global int4
typedef __attribute__((address_space(3))) v4i lv4i_t;   // LDS int4 (32-bit ptr)
static __device__ __forceinline__ void async_copy16(const void* g, void* l) {
    __builtin_amdgcn_global_load_async_to_lds_b128(
        (gv4i_t*)(unsigned long long)g,
        (lv4i_t*)(unsigned)(unsigned long long)l, 0, 0);
}
#else
#define HAS_ASYNC_LDS 0
#endif

// ---------------- gate: MLP -> softmax -> top1 + balance loss ----------------
__global__ void gate_kernel(const float* __restrict__ meta,
                            const float* __restrict__ w1, const float* __restrict__ b1,
                            const float* __restrict__ w2, const float* __restrict__ b2,
                            int* __restrict__ top1, float* __restrict__ loss_out) {
    __shared__ float probs[NB][NEXP];
    int b = threadIdx.x;
    if (b < NB) {
        float m9[9];
        #pragma unroll
        for (int k = 0; k < 9; ++k) m9[k] = meta[b * 9 + k];
        float logit[NEXP];
        #pragma unroll
        for (int e = 0; e < NEXP; ++e) logit[e] = b2[e];
        for (int j = 0; j < 128; ++j) {
            float h = b1[j];
            #pragma unroll
            for (int k = 0; k < 9; ++k) h += m9[k] * w1[k * 128 + j];
            h = fmaxf(h, 0.0f);
            #pragma unroll
            for (int e = 0; e < NEXP; ++e) logit[e] += h * w2[j * NEXP + e];
        }
        float mx = logit[0];
        #pragma unroll
        for (int e = 1; e < NEXP; ++e) mx = fmaxf(mx, logit[e]);
        float s = 0.0f, p[NEXP];
        #pragma unroll
        for (int e = 0; e < NEXP; ++e) { p[e] = __expf(logit[e] - mx); s += p[e]; }
        int best = 0; float bv = logit[0];
        #pragma unroll
        for (int e = 1; e < NEXP; ++e) if (logit[e] > bv) { bv = logit[e]; best = e; }
        top1[b] = best;
        #pragma unroll
        for (int e = 0; e < NEXP; ++e) probs[b][e] = p[e] / s;
    }
    __syncthreads();
    if (threadIdx.x == 0) {
        float se[NEXP]; float tot = 0.0f;
        #pragma unroll
        for (int e = 0; e < NEXP; ++e) {
            float a = 0.0f;
            for (int bb = 0; bb < NB; ++bb) a += probs[bb][e];
            se[e] = a; tot += a;
        }
        float mean = 0.0f, imp[NEXP];
        #pragma unroll
        for (int e = 0; e < NEXP; ++e) { imp[e] = se[e] / (tot + 1e-8f); mean += imp[e]; }
        mean *= (1.0f / NEXP);
        float var = 0.0f;
        #pragma unroll
        for (int e = 0; e < NEXP; ++e) { float d = imp[e] - mean; var += d * d; }
        var /= (NEXP - 1);              // ddof=1
        loss_out[0] = sqrtf(var);
    }
}

// ---------------- fold conv bias + BN into scale/bias ----------------
__global__ void bnprep_kernel(const float* __restrict__ cb, const float* __restrict__ g,
                              const float* __restrict__ be, const float* __restrict__ m,
                              const float* __restrict__ v,
                              float* __restrict__ s, float* __restrict__ t, int n) {
    int i = blockIdx.x * blockDim.x + threadIdx.x;
    if (i < n) {
        float sc = g[i] * rsqrtf(v[i] + 1e-5f);
        s[i] = sc;
        t[i] = (cb[i] - m[i]) * sc + be[i];
    }
}

// ---------------- weight transpose + f32->f16: [E][Cout][Cin][3][3] -> [E][tap][Cout][Cin] ----------------
__global__ void wprep_kernel(const float* __restrict__ w, _Float16* __restrict__ out) {
    int o = blockIdx.x * 256 + threadIdx.x;       // < 6*9*256*256
    int cin  = o & 255;
    int cout = (o >> 8) & 255;
    int r    = o >> 16;                           // e*9 + tap
    int tap  = r % 9;
    int e    = r / 9;
    int ky = tap / 3, kx = tap % 3;
    size_t src = ((((size_t)(e * 256 + cout)) * 256 + cin) * 3 + ky) * 3 + kx;
    out[o] = (_Float16)w[src];
}

// ---------------- implicit-GEMM conv3x3 + BN + ReLU, WMMA f16 ----------------
// LAYER==0: in f32 (d_in), out f16 (workspace).  LAYER==1: in f16, out f32 (d_out).
// K-steps s = cc*9 + tap (cc = cin chunk of 32, tap = 3x3 position).
// A (weights) double-buffered via async global->LDS; B is a raw halo tile
// [4 rows x 66 cols][32 cin] per chunk, shared by all 9 taps, double-buffered.
template <int LAYER>
__global__ __launch_bounds__(256) void conv_kernel(const void* __restrict__ inp,
                                                   const _Float16* __restrict__ wt,
                                                   const float* __restrict__ scale,
                                                   const float* __restrict__ bias,
                                                   const int* __restrict__ top1,
                                                   void* __restrict__ outp) {
    __shared__ __align__(16) _Float16 As[2][128 * A_STRIDE];
    __shared__ __align__(16) _Float16 Bs[2][B_POS * B_STRIDE];

    const int b      = blockIdx.z;
    const int e      = top1[b];
    const int m_base = blockIdx.y * 128;              // cout tile
    const int n_base = blockIdx.x * 128;              // pixel tile (2 image rows)
    const int y0     = n_base >> 6;                   // first image row of tile
    const int tid    = threadIdx.x;
    const int lane   = tid & 31;
    const int wave   = tid >> 5;
    const int wave_m = wave >> 2;                     // 0..1 -> 64 rows
    const int wave_n = wave & 3;                      // 0..3 -> 32 cols
    const int kh     = lane >> 4;
    const int lm     = lane & 15;

    const float*    inF = (const float*)inp    + (size_t)b * CCH * NPIX;
    const _Float16* inH = (const _Float16*)inp + (size_t)b * CCH * NPIX;
    const _Float16* wtE = wt + (size_t)e * 9 * 256 * 256;

    // A fill mapping (per thread): one cout row half
    const int am   = tid >> 1;
    const int acol = (tid & 1) * 16;
    // B fill mapping (per thread): 8 cin for a sweep of halo positions
    const int cgrp = tid & 3;
    const int pos0 = tid >> 2;

    v8f acc[4][2] = {};

    auto fillA = [&](int s, int buf) {
        const int tap = s % 9, cc = s / 9;
        const _Float16* g = wtE + ((size_t)tap * 256 + (m_base + am)) * 256 + cc * 32 + acol;
        _Float16* l = &As[buf][am * A_STRIDE + acol];
#if HAS_ASYNC_LDS
        async_copy16(g, l);
        async_copy16(g + 8, l + 8);
#else
        v8h w0 = *(const v8h*)g;
        v8h w1 = *(const v8h*)(g + 8);
        *(v8h*)l       = w0;
        *(v8h*)(l + 8) = w1;
#endif
    };

    auto fillB = [&](int cc, int buf) {
        const int cbase = cc * 32 + cgrp * 8;
        #pragma unroll
        for (int it = 0; it < 5; ++it) {
            const int pos = pos0 + it * 64;
            if (pos < B_POS) {
                const int lr = pos / 66;
                const int lc = pos - lr * 66;
                const int sy = y0 - 1 + lr;
                const int sx = lc - 1;
                const bool ok = ((unsigned)sy < (unsigned)IMGH) && ((unsigned)sx < (unsigned)IMGW);
                const int off = (sy << 6) + sx;
                v8h pk;
                #pragma unroll
                for (int j = 0; j < 8; ++j) {
                    _Float16 hv = (_Float16)0.0f;
                    if (ok) {
                        const size_t gi = ((size_t)(cbase + j) << 12) + off;
                        if (LAYER == 0) hv = (_Float16)inF[gi];
                        else            hv = inH[gi];
                    }
                    pk[j] = hv;
                }
                *(v8h*)&Bs[buf][pos * B_STRIDE + cgrp * 8] = pk;
            }
        }
    };

    // ---- prologue: stage step 0 ----
    fillA(0, 0);
    fillB(0, 0);
#if HAS_ASYNC_LDS
    __builtin_amdgcn_s_wait_asynccnt(0);
#endif
    __syncthreads();

    for (int s = 0; s < 72; ++s) {
        const int tap  = s % 9;
        const int cc   = s / 9;
        const int bufA = s & 1;
        const int bufB = cc & 1;

        // stage next step while computing this one (ping-pong buffers)
        if (s < 71) fillA(s + 1, bufA ^ 1);
        if (tap == 8 && cc < 7) fillB(cc + 1, bufB ^ 1);

        // ---- fragments (per-lane ISA layout: two 8-half groups) ----
        const _Float16* as_ = &As[bufA][0];
        const _Float16* bs_ = &Bs[bufB][0];
        const int kyr = tap / 3;      // ky + 1
        const int kxr = tap % 3;      // kx + 1

        v16h a[4], bq[2];
        #pragma unroll
        for (int mf = 0; mf < 4; ++mf) {
            const int m = wave_m * 64 + mf * 16 + lm;
            v8h lo = *(const v8h*)&as_[m * A_STRIDE + kh * 8];
            v8h hi = *(const v8h*)&as_[m * A_STRIDE + 16 + kh * 8];
            v16h f;
            #pragma unroll
            for (int i = 0; i < 8; ++i) { f[i] = lo[i]; f[i + 8] = hi[i]; }
            a[mf] = f;
        }
        #pragma unroll
        for (int nf = 0; nf < 2; ++nf) {
            const int n   = wave_n * 32 + nf * 16 + lm;
            const int pos = ((n >> 6) + kyr) * 66 + (n & 63) + kxr;
            v8h lo = *(const v8h*)&bs_[pos * B_STRIDE + kh * 8];
            v8h hi = *(const v8h*)&bs_[pos * B_STRIDE + 16 + kh * 8];
            v16h f;
            #pragma unroll
            for (int i = 0; i < 8; ++i) { f[i] = lo[i]; f[i + 8] = hi[i]; }
            bq[nf] = f;
        }

        // ---- 8 WMMAs per wave per K-step ----
        #pragma unroll
        for (int mf = 0; mf < 4; ++mf)
            #pragma unroll
            for (int nf = 0; nf < 2; ++nf)
                acc[mf][nf] = __builtin_amdgcn_wmma_f32_16x16x32_f16(
                    false, a[mf], false, bq[nf], (short)0, acc[mf][nf], false, false);

        if (s < 71) {
#if HAS_ASYNC_LDS
            __builtin_amdgcn_s_wait_asynccnt(2);   // step s+1's 2 copies may stay in flight
#endif
            __syncthreads();
        }
    }

    // ---- epilogue: fused BN scale/bias + ReLU ----
    _Float16* outH = (_Float16*)outp;
    float*    outF = (float*)outp;
    #pragma unroll
    for (int mf = 0; mf < 4; ++mf) {
        #pragma unroll
        for (int nf = 0; nf < 2; ++nf) {
            const int n   = wave_n * 32 + nf * 16 + lm;
            const int pix = n_base + n;
            #pragma unroll
            for (int r = 0; r < 8; ++r) {
                const int m    = wave_m * 64 + mf * 16 + r + 8 * kh;
                const int cout = m_base + m;
                const float sC = scale[e * 256 + cout];
                const float tC = bias[e * 256 + cout];
                float v = fmaxf(acc[mf][nf][r] * sC + tC, 0.0f);
                const size_t idx = ((size_t)b * 256 + cout) * NPIX + pix;
                if (LAYER == 0) outH[idx] = (_Float16)v;
                else            outF[idx] = v;
            }
        }
    }
}

// ---------------- host launch ----------------
extern "C" void kernel_launch(void* const* d_in, const int* in_sizes, int n_in,
                              void* d_out, int out_size, void* d_ws, size_t ws_size,
                              hipStream_t stream) {
    const float* x       = (const float*)d_in[0];
    const float* meta    = (const float*)d_in[1];
    const float* gw1     = (const float*)d_in[2];
    const float* gb1     = (const float*)d_in[3];
    const float* gw2     = (const float*)d_in[4];
    const float* gb2     = (const float*)d_in[5];
    const float* conv1_w = (const float*)d_in[6];
    const float* conv1_b = (const float*)d_in[7];
    const float* bn1_g   = (const float*)d_in[8];
    const float* bn1_b   = (const float*)d_in[9];
    const float* bn1_m   = (const float*)d_in[10];
    const float* bn1_v   = (const float*)d_in[11];
    const float* conv2_w = (const float*)d_in[12];
    const float* conv2_b = (const float*)d_in[13];
    const float* bn2_g   = (const float*)d_in[14];
    const float* bn2_b   = (const float*)d_in[15];
    const float* bn2_m   = (const float*)d_in[16];
    const float* bn2_v   = (const float*)d_in[17];

    char* ws = (char*)d_ws;
    int*   top1 = (int*)ws;                                // 64 B
    float* s1   = (float*)(ws + 256);
    float* t1   = s1 + NEXP * 256;
    float* s2   = t1 + NEXP * 256;
    float* t2   = s2 + NEXP * 256;                         // ends at 256+24576
    _Float16* wt1  = (_Float16*)(ws + 32768);
    _Float16* wt2  = wt1 + (size_t)WELEMS;
    _Float16* act1 = wt2 + (size_t)WELEMS;                 // 16*256*4096 halves

    float* out = (float*)d_out;

    gate_kernel<<<1, 32, 0, stream>>>(meta, gw1, gb1, gw2, gb2, top1, out + OUT_ELEMS);
    bnprep_kernel<<<6, 256, 0, stream>>>(conv1_b, bn1_g, bn1_b, bn1_m, bn1_v, s1, t1, NEXP * 256);
    bnprep_kernel<<<6, 256, 0, stream>>>(conv2_b, bn2_g, bn2_b, bn2_m, bn2_v, s2, t2, NEXP * 256);
    wprep_kernel<<<WELEMS / 256, 256, 0, stream>>>(conv1_w, wt1);
    wprep_kernel<<<WELEMS / 256, 256, 0, stream>>>(conv2_w, wt2);

    dim3 grid(32, 2, 16);   // pixel tiles x cout tiles x batch
    conv_kernel<0><<<grid, 256, 0, stream>>>(x,    wt1, s1, t1, top1, act1);
    conv_kernel<1><<<grid, 256, 0, stream>>>(act1, wt2, s2, t2, top1, out);
}
